// RandomTokenizer_88957362635159
// MI455X (gfx1250) — matrix operand order repaired and checked by reading, not compiled, not run on hardware
//
#include <hip/hip_runtime.h>
#include <hip/hip_bf16.h>

// ---------------------------------------------------------------------------
// MI455X plan:
//  - Fold post_w@pre_w -> Wc (256x768) so the selected-token path is ONE GEMM
//    (64 batches x [512x768]@[768x256]) done in bf16 WMMA w/ f32 accumulate
//    (v_wmma_f32_16x16x32_bf16). ~12.9 GFLOP.
//  - Everything else is bandwidth bound (~0.4 GB total traffic -> ~17us
//    at 23.3 TB/s), so keep those kernels simple and coalesced.
// ---------------------------------------------------------------------------

typedef __attribute__((ext_vector_type(16))) __bf16 v16bf;
typedef __attribute__((ext_vector_type(8)))  __bf16 v8bf;
typedef __attribute__((ext_vector_type(8)))  float  v8f;
typedef __attribute__((ext_vector_type(4)))  float  v4f;

#define BB   64
#define CIN  768
#define NTOK 1024
#define KSEL 512
#define ZD   256      // Z == DOUT == 256

__device__ __forceinline__ unsigned short f2bf(float x) {
    union { float f; unsigned int u; } c; c.f = x;
    unsigned int u = c.u;
    u += 0x7fffu + ((u >> 16) & 1u);       // round-to-nearest-even
    return (unsigned short)(u >> 16);
}

// --- 1. per-batch rank-based top-K + mask + per-batch min/max ---------------
__global__ void topk_kernel(const float* __restrict__ score,
                            float* __restrict__ mask, int* __restrict__ topk,
                            float* __restrict__ bmin, float* __restrict__ bmax) {
    __shared__ float s[NTOK];
    __shared__ float smn[NTOK];
    __shared__ float smx[NTOK];
    const int b = blockIdx.x, tid = threadIdx.x;
    const float mine = score[b * NTOK + tid];
    s[tid] = mine;
    __syncthreads();
    int rank = 0;
    #pragma unroll 8
    for (int j = 0; j < NTOK; ++j) {
        const float o = s[j];
        rank += (o > mine) || (o == mine && j < tid);
    }
    mask[b * NTOK + tid] = (rank < KSEL) ? 1.0f : 0.0f;
    if (rank < KSEL) topk[b * KSEL + rank] = tid;   // descending-score order
    smn[tid] = mine; smx[tid] = mine;
    __syncthreads();
    for (int st = NTOK / 2; st > 0; st >>= 1) {
        if (tid < st) {
            smn[tid] = fminf(smn[tid], smn[tid + st]);
            smx[tid] = fmaxf(smx[tid], smx[tid + st]);
        }
        __syncthreads();
    }
    if (tid == 0) { bmin[b] = smn[0]; bmax[b] = smx[0]; }
}

// --- 2. reduce 64 per-batch min/max -> smin, 1/max(smax-smin,eps) ----------
__global__ void minmax_reduce_kernel(const float* __restrict__ bmin,
                                     const float* __restrict__ bmax,
                                     float* __restrict__ scal) {
    __shared__ float mn[BB], mx[BB];
    const int t = threadIdx.x;
    mn[t] = bmin[t]; mx[t] = bmax[t];
    __syncthreads();
    for (int st = BB / 2; st > 0; st >>= 1) {
        if (t < st) { mn[t] = fminf(mn[t], mn[t + st]); mx[t] = fmaxf(mx[t], mx[t + st]); }
        __syncthreads();
    }
    if (t == 0) {
        scal[0] = mn[0];
        scal[1] = 1.0f / fmaxf(mx[0] - mn[0], 1e-5f);
    }
}

// --- 3a. fold Wc[o,c] = sum_z post_w[o,z]*pre_w[z,c] -> bf16 ----------------
__global__ void combine_w_kernel(const float* __restrict__ post_w,
                                 const float* __restrict__ pre_w,
                                 unsigned short* __restrict__ Wcb) {
    const int idx = blockIdx.x * blockDim.x + threadIdx.x;   // o*CIN + c
    const int o = idx / CIN, c = idx % CIN;
    float acc = 0.0f;
    #pragma unroll 8
    for (int z = 0; z < ZD; ++z)
        acc += post_w[o * ZD + z] * pre_w[z * CIN + c];
    Wcb[idx] = f2bf(acc);
}

// --- 3b. bias vectors: bc = post_w@pre_b + post_b ; cst = post_w@mask_token + post_b
__global__ void bias_kernel(const float* __restrict__ post_w,
                            const float* __restrict__ pre_b,
                            const float* __restrict__ mask_token,
                            const float* __restrict__ post_b,
                            float* __restrict__ bc, float* __restrict__ cst) {
    const int o = threadIdx.x;
    float ab = 0.0f, am = 0.0f;
    #pragma unroll 8
    for (int z = 0; z < ZD; ++z) {
        const float pw = post_w[o * ZD + z];
        ab += pw * pre_b[z];
        am += pw * mask_token[z];
    }
    bc[o]  = ab + post_b[o];
    cst[o] = am + post_b[o];
}

// --- 4. gather selected tokens + LayerNorm + bf16 pack ----------------------
// one wave32 per token: 24 channels/lane, single read of the data.
__global__ void gather_ln_kernel(const float* __restrict__ img,
                                 const int* __restrict__ topk,
                                 unsigned short* __restrict__ A) {
    const int wave = threadIdx.x >> 5, lane = threadIdx.x & 31;
    const int gidx = blockIdx.x * 8 + wave;        // 0 .. B*KSEL-1
    const int b = gidx >> 9, k = gidx & (KSEL - 1);
    const int t = topk[b * KSEL + k];
    const float* base = img + (size_t)b * CIN * NTOK + t;
    float v[24];
    float s = 0.0f, s2 = 0.0f;
    #pragma unroll
    for (int i = 0; i < 24; ++i) {
        v[i] = base[(size_t)(lane + 32 * i) * NTOK];
        s  += v[i];
        s2 += v[i] * v[i];
    }
    #pragma unroll
    for (int off = 16; off > 0; off >>= 1) {
        s  += __shfl_xor(s,  off, 32);
        s2 += __shfl_xor(s2, off, 32);
    }
    const float mu = s * (1.0f / CIN);
    const float var = s2 * (1.0f / CIN) - mu * mu;
    const float rs = rsqrtf(var + 1e-5f);
    unsigned short* dst = A + (size_t)gidx * CIN;
    #pragma unroll
    for (int i = 0; i < 24; ++i)
        dst[lane + 32 * i] = f2bf((v[i] - mu) * rs);
}

// --- 5. fill out with per-channel constant (non-selected columns) -----------
__global__ void fill_kernel(const float* __restrict__ cst, float* __restrict__ out) {
    const int i = blockIdx.x * blockDim.x + threadIdx.x;     // one v4f each
    const int base = i * 4;
    const int o = (base >> 10) & (ZD - 1);
    const float val = cst[o];
    v4f w = {val, val, val, val};
    *(v4f*)(out + base) = w;
}

// --- 6. folded GEMM via WMMA bf16: out[b, o, topk[k]] = A[k,:]@Wc[o,:] + bc[o]
// One 16x16 output tile per wave; 24 k-steps of v_wmma_f32_16x16x32_bf16.
__global__ void gemm_kernel(const unsigned short* __restrict__ A,
                            const unsigned short* __restrict__ Wc,
                            const float* __restrict__ bc,
                            const int* __restrict__ topk,
                            float* __restrict__ out) {
    const int wave = threadIdx.x >> 5, lane = threadIdx.x & 31;
    const int tile = blockIdx.x * 8 + wave;        // 0..511 per batch
    const int b  = blockIdx.y;
    const int k0 = (tile >> 4) * 16;               // token-row tile
    const int o0 = (tile & 15) * 16;               // out-channel tile
    const int lh = lane & 15;
    const bool hi = lane >= 16;

    // A fragment (16x32 bf16): lane row = k0+lh; lanes<16 carry K 0-15,
    // lanes>=16 carry K 16-31 -> two contiguous 16B chunks.
    const unsigned short* Arow = A + ((size_t)(b * KSEL + k0 + lh)) * CIN + (hi ? 16 : 0);
    // B fragment (32x16 bf16): lane col = o0+lh; 16 consecutive K per lane ->
    // one contiguous 32B chunk of the (row-major) Wc row.
    const unsigned short* Brow = Wc + (size_t)(o0 + lh) * CIN + (hi ? 16 : 0);

    v8f acc = {};
    for (int c0 = 0; c0 < CIN; c0 += 32) {
        __builtin_prefetch(Arow + c0 + 128, 0, 0);   // global_prefetch_b8
        const v8bf a0 = *(const v8bf*)(Arow + c0);
        const v8bf a1 = *(const v8bf*)(Arow + c0 + 8);
        const v16bf bf = *(const v16bf*)(Brow + c0);
        v16bf af;
        #pragma unroll
        for (int i = 0; i < 8; ++i) { af[i] = a0[i]; af[i + 8] = a1[i]; }
        acc = __builtin_amdgcn_wmma_f32_16x16x32_bf16(
            /*neg_a=*/false, af, /*neg_b=*/false, bf,
            /*c_mod=*/(short)0, acc, /*reuse_a=*/false, /*reuse_b=*/false);
    }

    // epilogue: D layout -> lane N = lane&15 (out channel), VGPR r -> M row
    const float bias = bc[o0 + lh];
    const int mbase = k0 + (hi ? 8 : 0);
    int tv[8];
    #pragma unroll
    for (int r = 0; r < 8; ++r) tv[r] = topk[b * KSEL + mbase + r];
    float* obase = out + ((size_t)b * ZD + o0 + lh) * NTOK;
    #pragma unroll
    for (int r = 0; r < 8; ++r) obase[tv[r]] = acc[r] + bias;
}

// --- 7. binary_map + score_map (nearest upsample x16), float4 stores --------
__global__ void maps_kernel(const float* __restrict__ mask,
                            const float* __restrict__ score,
                            const float* __restrict__ scal,
                            float* __restrict__ bmap, float* __restrict__ smap) {
    const int i = blockIdx.x * blockDim.x + threadIdx.x;     // one v4f each
    const int p = i * 4;                                     // pixel index
    const int b = p >> 18;                                   // 512*512 per batch
    const int y = (p >> 9) & 511;
    const int x = p & 511;
    const int n = (y >> 4) * 32 + (x >> 4);
    const float m  = mask[b * NTOK + n];
    const float sc = (score[b * NTOK + n] - scal[0]) * scal[1];
    v4f mv = {m, m, m, m};
    v4f sv = {sc, sc, sc, sc};
    *(v4f*)(bmap + p) = mv;
    *(v4f*)(smap + p) = sv;
}

extern "C" void kernel_launch(void* const* d_in, const int* in_sizes, int n_in,
                              void* d_out, int out_size, void* d_ws, size_t ws_size,
                              hipStream_t stream) {
    const float* img        = (const float*)d_in[0];   // (B, C, 32, 32)
    const float* score      = (const float*)d_in[1];   // (B, N)
    const float* pre_w      = (const float*)d_in[2];   // (Z, C)
    const float* pre_b      = (const float*)d_in[3];   // (Z,)
    const float* mask_token = (const float*)d_in[4];   // (1, DOUT, 1)
    const float* post_w     = (const float*)d_in[5];   // (DOUT, Z)
    const float* post_b     = (const float*)d_in[6];   // (DOUT,)

    float* out  = (float*)d_out;                       // (B, DOUT, 32, 32)
    float* bmap = out + 16777216;                      // (B, 1, 512, 512)
    float* smap = out + 33554432;                      // (B, 1, 512, 512)

    char* ws = (char*)d_ws;                            // all offsets 256B aligned
    int*            topk = (int*)(ws + 0);             // 64*512 ints   (128 KB)
    float*          mask = (float*)(ws + 131072);      // 64*1024 f32   (256 KB)
    float*          bmin = (float*)(ws + 393216);
    float*          bmax = (float*)(ws + 393472);
    float*          scal = (float*)(ws + 393728);      // smin, inv
    float*          bc   = (float*)(ws + 394240);      // 256 f32
    float*          cst  = (float*)(ws + 395264);      // 256 f32
    unsigned short* Wcb  = (unsigned short*)(ws + 396288);   // 256*768 bf16 (384 KB)
    unsigned short* Abf  = (unsigned short*)(ws + 790528);   // 64*512*768 bf16 (48 MB)

    topk_kernel<<<BB, NTOK, 0, stream>>>(score, mask, topk, bmin, bmax);
    minmax_reduce_kernel<<<1, BB, 0, stream>>>(bmin, bmax, scal);
    combine_w_kernel<<<(ZD * CIN) / 256, 256, 0, stream>>>(post_w, pre_w, Wcb);
    bias_kernel<<<1, ZD, 0, stream>>>(post_w, pre_b, mask_token, post_b, bc, cst);
    gather_ln_kernel<<<(BB * KSEL) / 8, 256, 0, stream>>>(img, topk, Abf);
    fill_kernel<<<16384, 256, 0, stream>>>(cst, out);
    gemm_kernel<<<dim3(64, BB), 256, 0, stream>>>(Abf, Wcb, bc, topk, out);
    maps_kernel<<<16384, 256, 0, stream>>>(mask, score, scal, bmap, smap);
}
